// TemporalGCN_53240414601591
// MI455X (gfx1250) — matrix operand order repaired and checked by reading, not compiled, not run on hardware
//
#include <hip/hip_runtime.h>
#include <hip/hip_bf16.h>
#include <math.h>

#define T_  20
#define B_  64
#define NN_ 400
#define N_  (B_*NN_)      /* 25600  */
#define E_  409600
#define H_  64

typedef __attribute__((ext_vector_type(16))) _Float16 v16h;
typedef __attribute__((ext_vector_type(8)))  _Float16 v8h;
typedef __attribute__((ext_vector_type(8)))  float    v8f;

// ---- WMMA fragment coordinates (CDNA5 ISA §7.12.2) ----
// 16-bit A (16x32): lane L holds row M=L%16; element i:
//   i in 0..7  -> K = (L/16)*8 + i
//   i in 8..15 -> K = 16 + (L/16)*8 + (i-8)
// => per lane the fragment is two contiguous 8xf16 (16B) chunks.
__device__ __forceinline__ void frag_coord(int lane, int i, int& m, int& k) {
    int v = i >> 1, h = i & 1;
    m = lane & 15;
    k = ((v >> 2) << 4) | (((lane >> 4) & 1) << 3) | ((v & 3) << 1) | h;
}
// Load a 16-f16 fragment from two contiguous 8-f16 chunks (p and p+16 halves).
__device__ __forceinline__ v16h load_frag(const _Float16* p) {
    v8h lo = *(const v8h*)p;
    v8h hi = *(const v8h*)(p + 16);
    return __builtin_shufflevector(lo, hi, 0,1,2,3,4,5,6,7,8,9,10,11,12,13,14,15);
}
__device__ __forceinline__ float sigm_(float x) { return 1.0f / (1.0f + __expf(-x)); }

// ---------------------------------------------------------------------------
// Weight prep: swizzle fused LSTM weights [W_ih^T ; W_hh^T] (128x256) into
// fragment-major f16 layout  wsw[((kb*16+ct)*32+lane)*16 + i]
// ---------------------------------------------------------------------------
__global__ void swizzle_lstm_k(const float* __restrict__ W_ih,
                               const float* __restrict__ W_hh,
                               _Float16* __restrict__ wsw) {
    int tid = blockIdx.x * 256 + threadIdx.x;           // 0..2047 = kb*16*32+ct*32+lane
    if (tid >= 4 * 16 * 32) return;
    int lane = tid & 31, ct = (tid >> 5) & 15, kb = tid >> 9;
    _Float16* o = wsw + (size_t)tid * 16;
    #pragma unroll
    for (int i = 0; i < 16; ++i) {
        int nn, k; frag_coord(lane, i, nn, k);
        int kg = kb * 32 + k;                            // 0..127  ([x|h] row)
        int col = ct * 16 + nn;                          // 0..255  (gate column)
        float v = (kg < 64) ? W_ih[col * 64 + kg] : W_hh[col * 64 + (kg - 64)];
        o[i] = (_Float16)v;
    }
}
__global__ void swizzle_w2_k(const float* __restrict__ W2, _Float16* __restrict__ w2s) {
    int tid = threadIdx.x;                               // 0..255 = kb*4*32+ct*32+lane
    int lane = tid & 31, ct = (tid >> 5) & 3, kb = tid >> 7;
    _Float16* o = w2s + (size_t)tid * 16;
    #pragma unroll
    for (int i = 0; i < 16; ++i) {
        int nn, k; frag_coord(lane, i, nn, k);
        o[i] = (_Float16)W2[(kb * 32 + k) * 64 + ct * 16 + nn];
    }
}
__global__ void bias_k(const float* __restrict__ b_ih, const float* __restrict__ b_hh,
                       float* __restrict__ bsc) {
    int c = threadIdx.x;
    bsc[c] = b_ih[c] + b_hh[c];
}

// ---------------------------------------------------------------------------
// Degree count (edges identical for both layers), then rsqrt.
// ---------------------------------------------------------------------------
__global__ void deg_count_k(const int* __restrict__ edges, float* __restrict__ deg) {
    long long g = (long long)blockIdx.x * blockDim.x + threadIdx.x;
    if (g >= (long long)T_ * E_) return;
    int t = (int)(g / E_), e = (int)(g % E_);
    int dst = edges[(size_t)t * 2 * E_ + E_ + e];
    atomicAdd(&deg[(size_t)t * N_ + dst], 1.0f);
}
__global__ void deg_to_dinv_k(float* __restrict__ d) {
    long long g = (long long)blockIdx.x * blockDim.x + threadIdx.x;
    if (g >= (long long)T_ * N_) return;
    d[g] = rsqrtf(d[g] + 1.0f);
}

// ---------------------------------------------------------------------------
// x @ W1  (IN_DIM=2 -> scalar FMA per node)
// ---------------------------------------------------------------------------
__global__ void xw1_k(const float* __restrict__ x, const float* __restrict__ W1,
                      float* __restrict__ xw) {
    long long g = (long long)blockIdx.x * blockDim.x + threadIdx.x;
    if (g >= (long long)T_ * N_) return;
    float x0 = x[g * 2], x1 = x[g * 2 + 1];
    float* o = xw + (size_t)g * 64;
    #pragma unroll 8
    for (int j = 0; j < 64; ++j) o[j] = x0 * W1[j] + x1 * W1[64 + j];
}

// ---------------------------------------------------------------------------
// Edge scatter: agg[dst] += xw[src] * dinv[src]*dinv[dst]
// ---------------------------------------------------------------------------
__global__ void scatter_k(const int* __restrict__ edges, const float* __restrict__ dinv,
                          const float* __restrict__ xw, float* __restrict__ agg) {
    long long g = (long long)blockIdx.x * blockDim.x + threadIdx.x;
    if (g >= (long long)T_ * E_ * 8) return;
    int part = (int)(g & 7);
    long long te = g >> 3;
    int t = (int)(te / E_), e = (int)(te % E_);
    const int* ep = edges + (size_t)t * 2 * E_;
    int src = ep[e], dst = ep[E_ + e];
    float nrm = dinv[(size_t)t * N_ + src] * dinv[(size_t)t * N_ + dst];
    const float* xs = xw  + ((size_t)t * N_ + src) * 64 + part * 8;
    float*       ad = agg + ((size_t)t * N_ + dst) * 64 + part * 8;
    #pragma unroll
    for (int j = 0; j < 8; ++j) atomicAdd(ad + j, xs[j] * nrm);
}

// ---------------------------------------------------------------------------
// Finalize GCN layer: self term + bias (+relu | +ego mask), cast -> f16
// ---------------------------------------------------------------------------
__global__ void finalize_k(const float* __restrict__ agg, const float* __restrict__ xw,
                           const float* __restrict__ dinv, const float* __restrict__ bias,
                           const unsigned char* __restrict__ ego, int do_relu,
                           _Float16* __restrict__ out) {
    long long g = (long long)blockIdx.x * blockDim.x + threadIdx.x;
    if (g >= (long long)T_ * N_ * 64) return;
    long long tn = g >> 6;
    int j = (int)(g & 63);
    float di = dinv[tn];
    float v = agg[g] + xw[g] * di * di + bias[j];
    if (do_relu) {
        v = fmaxf(v, 0.0f);
    } else {
        int t = (int)(tn / N_), n = (int)(tn % N_);
        int b = n / NN_, nn = n % NN_;
        if (ego[(size_t)b * T_ * NN_ + (size_t)t * NN_ + nn]) v = 0.0f;
    }
    out[g] = (_Float16)v;
}

// ---------------------------------------------------------------------------
// GCN layer-2 GEMM: xw2(512000x64) = h1(f16) @ W2(f16), f32 accumulate.
// 8 waves/block, one 16-row tile per wave; swizzled W2 fragments in LDS.
// ---------------------------------------------------------------------------
__global__ __launch_bounds__(256) void gcn2_wmma_k(const _Float16* __restrict__ hin,
                                                   const _Float16* __restrict__ w2s,
                                                   float* __restrict__ xout) {
    __shared__ __align__(32) _Float16 sW[2 * 4 * 32 * 16];   // 8 KB, fragment-major
    {
        float4* d = (float4*)sW; const float4* s = (const float4*)w2s;
        for (int i = threadIdx.x; i < 512; i += 256) d[i] = s[i];
    }
    __syncthreads();

    int wave = threadIdx.x >> 5, lane = threadIdx.x & 31;
    long long row0 = ((long long)blockIdx.x * 8 + wave) * 16;
    int n = lane & 15, hi = lane >> 4;

    const _Float16* ab = hin + ((size_t)row0 + (lane & 15)) * 64 + hi * 8;
    v16h a0 = load_frag(ab);
    v16h a1 = load_frag(ab + 32);

    #pragma unroll
    for (int ct = 0; ct < 4; ++ct) {
        v16h b0 = *(const v16h*)(sW + ((0 * 4 + ct) * 32 + lane) * 16);
        v16h b1 = *(const v16h*)(sW + ((1 * 4 + ct) * 32 + lane) * 16);
        v8f acc = {};
        acc = __builtin_amdgcn_wmma_f32_16x16x32_f16(false, a0, false, b0, (short)0, acc, false, false);
        acc = __builtin_amdgcn_wmma_f32_16x16x32_f16(false, a1, false, b1, (short)0, acc, false, false);
        #pragma unroll
        for (int r = 0; r < 8; ++r)
            xout[((size_t)row0 + r + hi * 8) * 64 + ct * 16 + n] = acc[r];
    }
}

// ---------------------------------------------------------------------------
// LSTM over T=20 + final FC(64->2).
// 4 waves/block, each wave owns 16 node rows for the whole sequence.
// gates(16x256) = [x_t | h](16x128) @ Wcat(128x256): 64 WMMA/step/wave.
// Weights staged to LDS with GLOBAL_LOAD_ASYNC_TO_LDS_B128 (ASYNCcnt DMA),
// h in per-wave LDS tile, cell state in VGPRs -> no barriers in time loop.
// ---------------------------------------------------------------------------
__global__ __launch_bounds__(128) void lstm_k(const _Float16* __restrict__ seq,
                                              const _Float16* __restrict__ wsw,
                                              const float* __restrict__ bias,
                                              const float* __restrict__ fcW,
                                              const float* __restrict__ fcb,
                                              float* __restrict__ out) {
    __shared__ __align__(32) _Float16 sW[4 * 16 * 32 * 16];  // 64 KB fragment-major
    __shared__ __align__(16) _Float16 sH[4][16 * 64];        // 8 KB per-wave h tiles

    // Async LDS DMA of the 64 KB weight block: each lane queues 32 B128
    // transfers (CDNA5 ISA, GV mode: 64-bit vaddr, no saddr).
    {
        unsigned int ldsa = (unsigned int)(size_t)(&sW[0]) + (unsigned int)threadIdx.x * 16u;
        unsigned long long ga = (unsigned long long)(size_t)wsw + (unsigned long long)threadIdx.x * 16ull;
        #pragma unroll
        for (int it = 0; it < 32; ++it) {
            asm volatile("global_load_async_to_lds_b128 %0, %1, off"
                         :: "v"(ldsa), "v"(ga) : "memory");
            ldsa += 128u * 16u;
            ga   += 128ull * 16ull;
        }
        asm volatile("s_wait_asynccnt 0" ::: "memory");
    }
    int wave = threadIdx.x >> 5, lane = threadIdx.x & 31;
    for (int i = lane; i < 16 * 64 / 8; i += 32) ((v8h*)sH[wave])[i] = (v8h)(_Float16)0.0f;
    __syncthreads();

    long long node0 = ((long long)blockIdx.x * 4 + wave) * 16;
    int n = lane & 15, hi = lane >> 4;

    v8f zero = {};
    v8f cst[4] = { zero, zero, zero, zero };   // cell state, 32 f32/lane

    for (int t = 0; t < T_; ++t) {
        // All four K-fragments of A = [x_t | h] loaded before any h update.
        const _Float16* ab = seq + ((size_t)t * N_ + (size_t)node0 + (lane & 15)) * 64 + hi * 8;
        const _Float16* hb = sH[wave] + (lane & 15) * 64 + hi * 8;
        v16h afr[4];
        afr[0] = load_frag(ab);
        afr[1] = load_frag(ab + 32);
        afr[2] = load_frag(hb);
        afr[3] = load_frag(hb + 32);

        #pragma unroll
        for (int jt = 0; jt < 4; ++jt) {           // hidden tile of 16 units
            v8f accg[4] = { zero, zero, zero, zero };   // i, f, g, o
            #pragma unroll
            for (int kb = 0; kb < 4; ++kb) {
                // 4 distinct B fragments (one per gate) -> one clause of 8
                // ds_load_b128, then 4 WMMAs reusing the same A operand.
                v16h bf0 = *(const v16h*)(sW + ((kb * 16 + (0 * 4 + jt)) * 32 + lane) * 16);
                v16h bf1 = *(const v16h*)(sW + ((kb * 16 + (1 * 4 + jt)) * 32 + lane) * 16);
                v16h bf2 = *(const v16h*)(sW + ((kb * 16 + (2 * 4 + jt)) * 32 + lane) * 16);
                v16h bf3 = *(const v16h*)(sW + ((kb * 16 + (3 * 4 + jt)) * 32 + lane) * 16);
                accg[0] = __builtin_amdgcn_wmma_f32_16x16x32_f16(false, afr[kb], false, bf0, (short)0, accg[0], false, false);
                accg[1] = __builtin_amdgcn_wmma_f32_16x16x32_f16(false, afr[kb], false, bf1, (short)0, accg[1], false, false);
                accg[2] = __builtin_amdgcn_wmma_f32_16x16x32_f16(false, afr[kb], false, bf2, (short)0, accg[2], false, false);
                accg[3] = __builtin_amdgcn_wmma_f32_16x16x32_f16(false, afr[kb], false, bf3, (short)0, accg[3], false, false);
            }
            int j = jt * 16 + n;
            float bi = bias[j], bfv = bias[64 + j], bg = bias[128 + j], bo = bias[192 + j];
            #pragma unroll
            for (int r = 0; r < 8; ++r) {
                float iv = sigm_(accg[0][r] + bi);
                float fv = sigm_(accg[1][r] + bfv);
                float gv = tanhf(accg[2][r] + bg);
                float ov = sigm_(accg[3][r] + bo);
                float c  = fv * cst[jt][r] + iv * gv;
                cst[jt][r] = c;
                float h  = ov * tanhf(c);
                sH[wave][(r + hi * 8) * 64 + j] = (_Float16)h;
            }
        }
    }
    // FC: 32 lanes -> 16 rows x 2 outputs, vector LDS reads
    int row = lane >> 1, k = lane & 1;
    float acc = fcb[k];
    const _Float16* hp = sH[wave] + row * 64;
    #pragma unroll
    for (int jc = 0; jc < 8; ++jc) {
        v8h hv = *(const v8h*)(hp + jc * 8);
        #pragma unroll
        for (int u = 0; u < 8; ++u)
            acc += (float)hv[u] * fcW[(jc * 8 + u) * 2 + k];
    }
    out[((size_t)node0 + row) * 2 + k] = acc;
}

// ---------------------------------------------------------------------------
extern "C" void kernel_launch(void* const* d_in, const int* in_sizes, int n_in,
                              void* d_out, int out_size, void* d_ws, size_t ws_size,
                              hipStream_t stream) {
    (void)in_sizes; (void)n_in; (void)out_size; (void)ws_size;
    const float* x     = (const float*)d_in[0];
    const int*   edges = (const int*)d_in[1];
    const unsigned char* ego = (const unsigned char*)d_in[2];
    const float* W1   = (const float*)d_in[3];
    const float* b1   = (const float*)d_in[4];
    const float* W2   = (const float*)d_in[5];
    const float* b2   = (const float*)d_in[6];
    const float* W_ih = (const float*)d_in[7];
    const float* W_hh = (const float*)d_in[8];
    const float* b_ih = (const float*)d_in[9];
    const float* b_hh = (const float*)d_in[10];
    const float* fcW  = (const float*)d_in[11];
    const float* fcb  = (const float*)d_in[12];
    float* out = (float*)d_out;

    const size_t TN   = (size_t)T_ * N_;       // 512,000
    const size_t TN64 = TN * 64;               // 32,768,000
    char* w = (char*)d_ws;
    const size_t OFF_DINV = 0;                         // TN f32      (< 2 MiB)
    const size_t OFF_XW   = 2u * 1024 * 1024;          // TN64 f32    (125 MiB)
    const size_t OFF_AGG  = OFF_XW  + TN64 * 4;        // TN64 f32    (125 MiB)
    const size_t OFF_HSEQ = OFF_AGG + TN64 * 4;        // TN64 f16    (62.5 MiB)
    const size_t OFF_WSW  = OFF_HSEQ + TN64 * 2;       // 4*16*32*16 f16 (64 KB)
    const size_t OFF_W2S  = OFF_WSW + 4*16*32*16 * 2;  // 2*4*32*16  f16 (8 KB)
    const size_t OFF_BIAS = OFF_W2S + 2*4*32*16 * 2;   // 256 f32
    float*    dinv = (float*)(w + OFF_DINV);
    float*    xw   = (float*)(w + OFF_XW);
    float*    agg  = (float*)(w + OFF_AGG);
    _Float16* hseq = (_Float16*)(w + OFF_HSEQ);
    _Float16* wsw  = (_Float16*)(w + OFF_WSW);
    _Float16* w2s  = (_Float16*)(w + OFF_W2S);
    float*    bsc  = (float*)(w + OFF_BIAS);

    hipMemsetAsync(dinv, 0, TN * 4, stream);
    hipMemsetAsync(agg,  0, TN64 * 4, stream);
    swizzle_lstm_k<<<8, 256, 0, stream>>>(W_ih, W_hh, wsw);
    swizzle_w2_k<<<1, 256, 0, stream>>>(W2, w2s);
    bias_k<<<1, 256, 0, stream>>>(b_ih, b_hh, bsc);

    // degrees (shared by both layers)
    {
        long long nthr = (long long)T_ * E_;
        deg_count_k<<<(unsigned)((nthr + 255) / 256), 256, 0, stream>>>(edges, dinv);
        deg_to_dinv_k<<<(unsigned)((TN + 255) / 256), 256, 0, stream>>>(dinv);
    }
    // layer 1
    xw1_k<<<(unsigned)((TN + 255) / 256), 256, 0, stream>>>(x, W1, xw);
    {
        long long nthr = (long long)T_ * E_ * 8;
        scatter_k<<<(unsigned)((nthr + 255) / 256), 256, 0, stream>>>(edges, dinv, xw, agg);
    }
    finalize_k<<<(unsigned)((TN64 + 255) / 256), 256, 0, stream>>>(agg, xw, dinv, b1, ego, 1, hseq);
    // layer 2
    gcn2_wmma_k<<<(unsigned)(TN / 16 / 8), 256, 0, stream>>>(hseq, w2s, xw);
    hipMemsetAsync(agg, 0, TN64 * 4, stream);
    {
        long long nthr = (long long)T_ * E_ * 8;
        scatter_k<<<(unsigned)((nthr + 255) / 256), 256, 0, stream>>>(edges, dinv, xw, agg);
    }
    finalize_k<<<(unsigned)((TN64 + 255) / 256), 256, 0, stream>>>(agg, xw, dinv, b2, ego, 0, hseq);
    // LSTM + FC
    lstm_k<<<N_ / 64, 128, 0, stream>>>(hseq, wsw, bsc, fcW, fcb, out);
}
